// SelfAttention2D_13228499272031
// MI455X (gfx1250) — matrix operand order repaired and checked by reading, not compiled, not run on hardware
//
#include <hip/hip_runtime.h>

static constexpr int Bb  = 4;
static constexpr int Cc  = 256;
static constexpr int Nn  = 4096;   // H*W
static constexpr int Dqk = 32;     // CQK
static constexpr int JT  = 128;    // key/value tile (columns per softmax round)
static constexpr float LOG2E = 1.44269504088896340736f;

typedef __attribute__((ext_vector_type(16))) __bf16 bf16x16;
typedef __attribute__((ext_vector_type(8)))  float  floatx8;

__device__ __forceinline__ unsigned short f32_to_bf16(float f) {
  union { float f; unsigned int u; } a; a.f = f;
  unsigned int r = a.u + 0x7FFFu + ((a.u >> 16) & 1u);   // round-to-nearest-even
  return (unsigned short)(r >> 16);
}

__device__ __forceinline__ bf16x16 make_frag(uint4 lo, uint4 hi) {
  union { uint4 q[2]; bf16x16 v; } u;
  u.q[0] = lo; u.q[1] = hi;
  return u.v;
}

// A fragment (16x32 bf16, row-major rows of length ld).
// ISA layout: lanes 0-15 row M=lane, K 0..7 (v0-3) & 16..23 (v4-7);
//             lanes 16-31 same rows, K 8..15 & 24..31.
__device__ __forceinline__ bf16x16 load_frag_a(const unsigned short* base, int ld,
                                               int lm, int half) {
  const unsigned short* p = base + lm * ld + half * 8;
  return make_frag(*(const uint4*)p, *(const uint4*)(p + 16));
}

// B fragment (32x16 bf16, column-major: base points to [col][k], rows of length ld).
// ISA layout: lanes 0-15 col N=lane, K 0..7 & 16..23; lanes 16-31 col N=lane-16, K 8..15 & 24..31.
__device__ __forceinline__ bf16x16 load_frag_b(const unsigned short* base, int ld,
                                               int lm, int half) {
  const unsigned short* p = base + lm * ld + half * 8;
  return make_frag(*(const uint4*)p, *(const uint4*)(p + 16));
}

// ---------------- Projections ----------------
// Q is pre-scaled by log2(e) so the attention kernel can use pure exp2.

__global__ __launch_bounds__(256) void qk_proj_kernel(
    const float* __restrict__ x, const float* __restrict__ wq, const float* __restrict__ bq,
    const float* __restrict__ wk, const float* __restrict__ bk,
    unsigned short* __restrict__ Qh, unsigned short* __restrict__ Kh) {
  int tid = blockIdx.x * 256 + threadIdx.x;     // Bb*Nn threads
  int n = tid & (Nn - 1);
  int b = tid >> 12;
  const float* xb = x + (size_t)b * Cc * Nn + n;
  unsigned short* qo = Qh + ((size_t)b * Nn + n) * Dqk;
  unsigned short* ko = Kh + ((size_t)b * Nn + n) * Dqk;
  for (int d0 = 0; d0 < Dqk; d0 += 8) {
    float aq[8], ak[8];
#pragma unroll
    for (int u = 0; u < 8; ++u) { aq[u] = bq[d0 + u]; ak[u] = bk[d0 + u]; }
    for (int c = 0; c < Cc; ++c) {
      float xv = xb[(size_t)c * Nn];
#pragma unroll
      for (int u = 0; u < 8; ++u) {
        aq[u] = fmaf(wq[(d0 + u) * Cc + c], xv, aq[u]);
        ak[u] = fmaf(wk[(d0 + u) * Cc + c], xv, ak[u]);
      }
    }
#pragma unroll
    for (int u = 0; u < 8; ++u) {
      qo[d0 + u] = f32_to_bf16(aq[u] * LOG2E);   // fold log2(e) into Q
      ko[d0 + u] = f32_to_bf16(ak[u]);
    }
  }
}

__global__ __launch_bounds__(256) void v_proj_kernel(
    const float* __restrict__ x, const float* __restrict__ wv, const float* __restrict__ bv,
    unsigned short* __restrict__ Vh) {
  // grid = Bb * (Cc/16) * (Nn/256)
  int nblk = blockIdx.x & 15;
  int cg   = (blockIdx.x >> 4) & 15;
  int b    = blockIdx.x >> 8;
  int n  = nblk * 256 + threadIdx.x;
  int c0 = cg * 16;
  const float* xb = x + (size_t)b * Cc * Nn + n;
  float a[16];
#pragma unroll
  for (int u = 0; u < 16; ++u) a[u] = bv[c0 + u];
  for (int c = 0; c < Cc; ++c) {
    float xv = xb[(size_t)c * Nn];
#pragma unroll
    for (int u = 0; u < 16; ++u)
      a[u] = fmaf(wv[(c0 + u) * Cc + c], xv, a[u]);
  }
#pragma unroll
  for (int u = 0; u < 16; ++u)
    Vh[((size_t)b * Cc + c0 + u) * Nn + n] = f32_to_bf16(a[u]);
}

// ---------------- Flash attention (bf16 WMMA, f32 accum, base-2 softmax) -----
// Block = 256 threads (8 waves) handles one (batch, 64-query tile).
// Wave w: query rows iw..iw+15 (w>>1), channels (w&1)*128..+127.

__global__ __launch_bounds__(256) void attn_kernel(
    const unsigned short* __restrict__ Qh,
    const unsigned short* __restrict__ Kh,
    const unsigned short* __restrict__ Vh,
    const float* __restrict__ x,
    const float* __restrict__ gamma,
    float* __restrict__ out) {
  __shared__ unsigned short p_lds[8][16 * JT];   // per-wave P tile (16 x 128), 32KB

  const int b    = blockIdx.x >> 6;
  const int i0   = (blockIdx.x & 63) * 64;
  const int wave = threadIdx.x >> 5;
  const int lane = threadIdx.x & 31;
  const int lm   = lane & 15;
  const int half = lane >> 4;
  const int iw   = i0 + (wave >> 1) * 16;
  const int c0w  = (wave & 1) * 128;

  const unsigned short* Qb = Qh + ((size_t)b * Nn + iw) * Dqk;
  const unsigned short* Kb = Kh + (size_t)b * Nn * Dqk;
  const unsigned short* Vb = Vh + (size_t)b * Cc * Nn;

  bf16x16 qf = load_frag_a(Qb, Dqk, lm, half);   // Q tile, loaded once

  floatx8 acc[8];                 // 16 rows x 128 ch (8 WMMA D tiles)
  float mrow[8], lrow[8];         // per D-vgpr row stats (rows r / r+8 per half)
  floatx8 zero8;
#pragma unroll
  for (int r = 0; r < 8; ++r) zero8[r] = 0.0f;
#pragma unroll
  for (int t = 0; t < 8; ++t) acc[t] = zero8;
#pragma unroll
  for (int r = 0; r < 8; ++r) { mrow[r] = -1e30f; lrow[r] = 0.0f; }

  unsigned short* pw = &p_lds[wave][0];

  for (int j0 = 0; j0 < Nn; j0 += JT) {
    // S = Q^T K  (8 tiles of 16x16, K-dim = 32 in one WMMA each)
    floatx8 S[8];
#pragma unroll
    for (int t = 0; t < 8; ++t) {
      bf16x16 kf = load_frag_b(Kb + (size_t)(j0 + t * 16) * Dqk, Dqk, lm, half);
      S[t] = __builtin_amdgcn_wmma_f32_16x16x32_bf16(false, qf, false, kf,
                                                     (short)0, zero8, false, false);
    }

    // online softmax (base 2): new row max (tiles + 16-lane half butterfly)
    float mnew[8], alpha[8], rs[8];
#pragma unroll
    for (int r = 0; r < 8; ++r) {
      float v = fmaxf(fmaxf(fmaxf(S[0][r], S[1][r]), fmaxf(S[2][r], S[3][r])),
                      fmaxf(fmaxf(S[4][r], S[5][r]), fmaxf(S[6][r], S[7][r])));
#pragma unroll
      for (int msk = 1; msk <= 8; msk <<= 1)
        v = fmaxf(v, __shfl_xor(v, msk, 32));
      mnew[r]  = fmaxf(mrow[r], v);
      alpha[r] = __builtin_amdgcn_exp2f(mrow[r] - mnew[r]);
      mrow[r]  = mnew[r];
      rs[r]    = 0.0f;
    }

    // P = 2^(S - m), write bf16 tile to LDS in row-major [16][JT]
#pragma unroll
    for (int t = 0; t < 8; ++t)
#pragma unroll
      for (int r = 0; r < 8; ++r) {
        float p = __builtin_amdgcn_exp2f(S[t][r] - mnew[r]);
        rs[r] += p;
        pw[(r + half * 8) * JT + t * 16 + lm] = f32_to_bf16(p);
      }

#pragma unroll
    for (int r = 0; r < 8; ++r) {
      float v = rs[r];
#pragma unroll
      for (int msk = 1; msk <= 8; msk <<= 1)
        v += __shfl_xor(v, msk, 32);
      lrow[r] = lrow[r] * alpha[r] + v;
    }

    // rescale running accumulators
#pragma unroll
    for (int t = 0; t < 8; ++t)
#pragma unroll
      for (int r = 0; r < 8; ++r)
        acc[t][r] *= alpha[r];

    asm volatile("s_wait_dscnt 0" ::: "memory");   // P stores visible to this wave

    // O += P @ V^T in 4 passes of 8 independent WMMAs (shared A per pass)
#pragma unroll
    for (int pass = 0; pass < 4; ++pass) {
      bf16x16 pf = load_frag_a(pw + pass * 32, JT, lm, half);  // K = 32*pass..+31
#pragma unroll
      for (int ct = 0; ct < 8; ++ct) {
        const unsigned short* vb =
            Vb + (size_t)(c0w + ct * 16) * Nn + j0 + pass * 32;
        bf16x16 vf = load_frag_b(vb, Nn, lm, half);
        acc[ct] = __builtin_amdgcn_wmma_f32_16x16x32_bf16(false, pf, false, vf,
                                                          (short)0, acc[ct],
                                                          false, false);
      }
    }
  }

  // epilogue: out = gamma * (acc * 1/l) + x
  float g = gamma[0];
  float rinv[8];
#pragma unroll
  for (int r = 0; r < 8; ++r) rinv[r] = __builtin_amdgcn_rcpf(lrow[r]);
#pragma unroll
  for (int ct = 0; ct < 8; ++ct) {
    int c = c0w + ct * 16 + lm;
#pragma unroll
    for (int r = 0; r < 8; ++r) {
      int i = iw + r + half * 8;
      size_t idx = ((size_t)b * Cc + c) * Nn + i;
      out[idx] = g * (acc[ct][r] * rinv[r]) + x[idx];
    }
  }
}

// ---------------- Launch ----------------

extern "C" void kernel_launch(void* const* d_in, const int* in_sizes, int n_in,
                              void* d_out, int out_size, void* d_ws, size_t ws_size,
                              hipStream_t stream) {
  const float* x     = (const float*)d_in[0];
  const float* wq    = (const float*)d_in[1];
  const float* bq    = (const float*)d_in[2];
  const float* wk    = (const float*)d_in[3];
  const float* bk    = (const float*)d_in[4];
  const float* wv    = (const float*)d_in[5];
  const float* bv    = (const float*)d_in[6];
  const float* gamma = (const float*)d_in[7];
  float* out = (float*)d_out;

  // workspace: Qh (1MB) | Kh (1MB) | Vh (8MB), all bf16
  unsigned short* Qh = (unsigned short*)d_ws;
  unsigned short* Kh = Qh + (size_t)Bb * Nn * Dqk;
  unsigned short* Vh = Kh + (size_t)Bb * Nn * Dqk;

  qk_proj_kernel<<<(Bb * Nn) / 256, 256, 0, stream>>>(x, wq, bq, wk, bk, Qh, Kh);
  v_proj_kernel<<<Bb * (Cc / 16) * (Nn / 256), 256, 0, stream>>>(x, wv, bv, Vh);
  attn_kernel<<<Bb * (Nn / 64), 256, 0, stream>>>(Qh, Kh, Vh, x, gamma, out);
}
